// MultiHeadSelfAttention_44727789421362
// MI455X (gfx1250) — compile-verified, matrix operand order checked by reading, not run on hardware
//
#include <hip/hip_runtime.h>
#include <hip/hip_bf16.h>
#include <stdint.h>

#define DI __device__ __forceinline__

typedef __attribute__((ext_vector_type(16))) __bf16 v16bf;
typedef __attribute__((ext_vector_type(8)))  __bf16 v8bf;
typedef __attribute__((ext_vector_type(8)))  float  v8f;
typedef int v4i_gcc __attribute__((__vector_size__(16)));  // matches builtin's V4i

constexpr int kB   = 2;
constexpr int kS   = 2048;
constexpr int kD   = 2048;
constexpr int kH   = 32;
constexpr int kKVH = 8;
constexpr int kHD  = 64;
constexpr int kQO  = kH * kHD;        // 2048
constexpr int kKVO = kKVH * kHD;      // 512
constexpr int kOP  = kQO + 2 * kKVO;  // 3072
constexpr int kM   = kB * kS;         // 4096

// ---------------- bf16 helpers ----------------
DI __bf16 f2bf(float f) {
  unsigned u = __builtin_bit_cast(unsigned, f);
  u += 0x7FFFu + ((u >> 16) & 1u);              // round-to-nearest-even
  unsigned short s = (unsigned short)(u >> 16);
  return __builtin_bit_cast(__bf16, s);
}

// Load one 16-bit A/B WMMA fragment from (row-major, K-contiguous) storage.
// Per ISA layout: per lane, halves 0..7 come from cols [c..c+7], halves 8..15
// from cols [c+16..c+23], where c = (lane>>4)*8 has been folded into `p`.
DI v16bf load_frag16(const __bf16* p) {
  v8bf lo = *(const v8bf*)(p);
  v8bf hh = *(const v8bf*)(p + 16);
  return __builtin_shufflevector(lo, hh, 0,1,2,3,4,5,6,7,8,9,10,11,12,13,14,15);
}

DI v8f v8f_zero() { v8f z = {0.f,0.f,0.f,0.f,0.f,0.f,0.f,0.f}; return z; }

DI v8f wmma_bf16(v16bf a, v16bf b, v8f c) {
  return __builtin_amdgcn_wmma_f32_16x16x32_bf16(false, a, false, b,
                                                 (short)0, c, false, false);
}

// ---------------- async global->LDS copy (CDNA5), guarded ----------------
#if defined(__has_builtin)
#  if __has_builtin(__builtin_amdgcn_global_load_async_to_lds_b128) && \
      __has_builtin(__builtin_amdgcn_s_wait_asynccnt)
#    define USE_ASYNC_COPY 1
#  endif
#endif
#ifndef USE_ASYNC_COPY
#  define USE_ASYNC_COPY 0
#endif

DI void copy16_to_lds(void* lds_dst, const void* gsrc) {
#if USE_ASYNC_COPY
  __builtin_amdgcn_global_load_async_to_lds_b128(
      (__attribute__((address_space(1))) v4i_gcc*)(unsigned long long)gsrc,
      (__attribute__((address_space(3))) v4i_gcc*)(unsigned)(unsigned long long)lds_dst,
      0, 0);
#else
  *(uint4*)lds_dst = *(const uint4*)gsrc;
#endif
}

DI void wait_async_copies() {
#if USE_ASYNC_COPY
  __builtin_amdgcn_s_wait_asynccnt(0);
#endif
}

// ---------------- fp32 -> bf16 conversion ----------------
__global__ __launch_bounds__(256) void cvt_bf16(const float* __restrict__ in,
                                                __bf16* __restrict__ out, int n) {
  int i = (blockIdx.x * 256 + threadIdx.x) * 8;
  if (i >= n) return;
  float4 a  = *(const float4*)(in + i);
  float4 b2 = *(const float4*)(in + i + 4);
  v8bf o;
  o[0] = f2bf(a.x);  o[1] = f2bf(a.y);  o[2] = f2bf(a.z);  o[3] = f2bf(a.w);
  o[4] = f2bf(b2.x); o[5] = f2bf(b2.y); o[6] = f2bf(b2.z); o[7] = f2bf(b2.w);
  *(v8bf*)(out + i) = o;
}

// ---------------- C(MxN,f32) = A(MxK,bf16) * B(NxK,bf16)^T ----------------
// Block: 256 thr (8 waves as 2x4). Block tile 128x128, wave tile 64x32, BK=32.
__global__ __launch_bounds__(256) void gemm_bf16_nt(
    const __bf16* __restrict__ A, const __bf16* __restrict__ Bw,
    float* __restrict__ C, int M, int N, int K) {
  constexpr int BM = 128, BN = 128, BK = 32;
  __shared__ __align__(16) __bf16 As[2][BM * BK];
  __shared__ __align__(16) __bf16 Bs[2][BN * BK];
  (void)M;
  const int bm = blockIdx.y * BM, bn = blockIdx.x * BN;
  const int tid = threadIdx.x, wid = tid >> 5, lane = tid & 31;
  const int l16 = lane & 15, hiq = lane >> 4;
  const int wm = (wid >> 2) * 64, wn = (wid & 3) * 32;

  v8f acc[4][2];
#pragma unroll
  for (int i = 0; i < 4; i++)
#pragma unroll
    for (int j = 0; j < 2; j++) acc[i][j] = v8f_zero();

  // prologue: stage k-block 0 into buffer 0
#pragma unroll
  for (int i = 0; i < 2; i++) {
    int c = tid + i * 256, row = c >> 2, col = (c & 3) * 8;
    copy16_to_lds(&As[0][row * BK + col], A + (size_t)(bm + row) * K + col);
    copy16_to_lds(&Bs[0][row * BK + col], Bw + (size_t)(bn + row) * K + col);
  }
  wait_async_copies();
  __syncthreads();

  const int nk = K / BK;
  for (int kb = 0; kb < nk; ++kb) {
    const int cur = kb & 1, nxt = cur ^ 1;
    if (kb + 1 < nk) {
      const int k0 = (kb + 1) * BK;
#pragma unroll
      for (int i = 0; i < 2; i++) {
        int c = tid + i * 256, row = c >> 2, col = (c & 3) * 8;
        copy16_to_lds(&As[nxt][row * BK + col], A + (size_t)(bm + row) * K + k0 + col);
        copy16_to_lds(&Bs[nxt][row * BK + col], Bw + (size_t)(bn + row) * K + k0 + col);
      }
    }
    v16bf bfr[2];
#pragma unroll
    for (int nf = 0; nf < 2; nf++)
      bfr[nf] = load_frag16(&Bs[cur][(wn + nf * 16 + l16) * BK + hiq * 8]);
#pragma unroll
    for (int mf = 0; mf < 4; mf++) {
      v16bf afr = load_frag16(&As[cur][(wm + mf * 16 + l16) * BK + hiq * 8]);
#pragma unroll
      for (int nf = 0; nf < 2; nf++)
        acc[mf][nf] = wmma_bf16(afr, bfr[nf], acc[mf][nf]);
    }
    wait_async_copies();
    __syncthreads();
  }

  // epilogue: C-layout -> global (lane l16 = col, VGPR r = row r / r+8)
#pragma unroll
  for (int mf = 0; mf < 4; mf++)
#pragma unroll
    for (int nf = 0; nf < 2; nf++) {
      const int col = bn + wn + nf * 16 + l16;
#pragma unroll
      for (int r = 0; r < 8; r++) {
        const int row = bm + wm + mf * 16 + hiq * 8 + r;
        C[(size_t)row * N + col] = acc[mf][nf][r];
      }
    }
}

// ---------------- RoPE + RMSNorm + V scale/emit ----------------
// One block per (b,s). One wave handles one 64-dim head: lane owns pair
// (d=lane, d=lane+32) which is exactly the RoPE rotation pair.
__global__ __launch_bounds__(256) void rope_norm(
    const float* __restrict__ proj, const float* __restrict__ gain,
    __bf16* __restrict__ qn, __bf16* __restrict__ kn,
    __bf16* __restrict__ vbo, float* __restrict__ vflat) {
  const int m = blockIdx.x;
  const int b = m / kS, s = m % kS;
  const int wid = threadIdx.x >> 5, lane = threadIdx.x & 31;
  const float* row = proj + (size_t)m * kOP;
  const float g = gain[0];
  const float inv_freq = __expf(-((float)(2 * lane) / (float)kHD) * 9.210340371976184f);
  float sn, cs;
  __sincosf((float)s * inv_freq, &sn, &cs);

  // Q: 32 heads over 8 waves
#pragma unroll
  for (int it = 0; it < 4; ++it) {
    const int hq = wid + it * 8;
    float t1 = row[hq * kHD + lane], t2 = row[hq * kHD + 32 + lane];
    float o1 = t1 * cs - t2 * sn, o2 = t2 * cs + t1 * sn;
    float ss = o1 * o1 + o2 * o2;
#pragma unroll
    for (int off = 16; off >= 1; off >>= 1) ss += __shfl_xor(ss, off, 32);
    const float inv = rsqrtf(ss * (1.f / kHD) + 1e-6f) * g;
    qn[(size_t)m * kD + hq * kHD + lane]      = f2bf(o1 * inv);
    qn[(size_t)m * kD + hq * kHD + 32 + lane] = f2bf(o2 * inv);
  }
  // K: 8 kv heads, one per wave (rmsnorm commutes with head-repeat)
  {
    const int hk = wid;
    float t1 = row[kQO + hk * kHD + lane], t2 = row[kQO + hk * kHD + 32 + lane];
    float o1 = t1 * cs - t2 * sn, o2 = t2 * cs + t1 * sn;
    float ss = o1 * o1 + o2 * o2;
#pragma unroll
    for (int off = 16; off >= 1; off >>= 1) ss += __shfl_xor(ss, off, 32);
    const float inv = rsqrtf(ss * (1.f / kHD) + 1e-6f);
    kn[(size_t)m * kKVO + hk * kHD + lane]      = f2bf(o1 * inv);
    kn[(size_t)m * kKVO + hk * kHD + 32 + lane] = f2bf(o2 * inv);
  }
  // V: scale, store bf16 for P*V, and emit V_flat output (4x head repeat)
  {
    const int hv = wid;
    const float vsc = rsqrtf(1.f + 1e-8f);
    float v1 = row[kQO + kKVO + hv * kHD + lane] * vsc;
    float v2 = row[kQO + kKVO + hv * kHD + 32 + lane] * vsc;
    vbo[(size_t)m * kKVO + hv * kHD + lane]      = f2bf(v1);
    vbo[(size_t)m * kKVO + hv * kHD + 32 + lane] = f2bf(v2);
#pragma unroll
    for (int rph = 0; rph < 4; ++rph) {
      const int hh = hv * 4 + rph;
      float* o = vflat + ((size_t)(hh * kB + b) * kS + s) * kHD;
      o[lane] = v1;
      o[32 + lane] = v2;
    }
  }
}

// ---------------- flash attention (causal) ----------------
// Grid (S/256, H, B); block 256 thr; wave w owns q rows [qb*256+w*32, +32).
__global__ __launch_bounds__(256) void attn_fwd(
    const __bf16* __restrict__ qn, const __bf16* __restrict__ kn,
    const __bf16* __restrict__ vb, __bf16* __restrict__ attnb) {
  __shared__ __align__(16) __bf16 Ks[64 * 64];      // (key, hd)
  __shared__ __align__(16) __bf16 Vt[64 * 64];      // (hd, key)
  __shared__ __align__(16) __bf16 Ps[8][32 * 64];   // per-wave P staging (q, key)
  const int qb = blockIdx.x, h = blockIdx.y, b = blockIdx.z;
  const int kvh = h >> 2;  // H/KVH = 4
  const int tid = threadIdx.x, wid = tid >> 5, lane = tid & 31;
  const int l16 = lane & 15, hiq = lane >> 4;
  const int q0 = qb * 256 + wid * 32;

  // resident Q A-fragments: [mfrag][k-chunk]
  v16bf qf[2][2];
#pragma unroll
  for (int mf = 0; mf < 2; mf++)
#pragma unroll
    for (int kc = 0; kc < 2; kc++)
      qf[mf][kc] = load_frag16(
          qn + (size_t)(b * kS + q0 + mf * 16 + l16) * kD + h * kHD + kc * 32 + hiq * 8);

  v8f ov[2][4];
  float mrow[2][8], lrow[2][8];
#pragma unroll
  for (int mf = 0; mf < 2; mf++) {
#pragma unroll
    for (int nf = 0; nf < 4; nf++) ov[mf][nf] = v8f_zero();
#pragma unroll
    for (int r = 0; r < 8; r++) { mrow[mf][r] = -3.0e38f; lrow[mf][r] = 0.f; }
  }

  const int nkt = (qb + 1) * 4;  // k-tiles of 64 needed for causal coverage
  for (int kt = 0; kt < nkt; ++kt) {
    const int kbase = kt * 64;
    // stage K tile (64 keys x 64 hd), async path when available
#pragma unroll
    for (int i = 0; i < 2; i++) {
      int c = tid + i * 256, row = c >> 3, col = (c & 7) * 8;
      copy16_to_lds(&Ks[row * 64 + col],
                    kn + (size_t)(b * kS + kbase + row) * kKVO + kvh * kHD + col);
    }
    // stage V transposed (hd, key): coalesced 32B read, strided LDS writes
    {
      const int key = tid & 63, seg = tid >> 6, hd0 = seg * 16;
      const __bf16* src = vb + (size_t)(b * kS + kbase + key) * kKVO + kvh * kHD + hd0;
      v8bf a0 = *(const v8bf*)src;
      v8bf a1 = *(const v8bf*)(src + 8);
#pragma unroll
      for (int j = 0; j < 8; j++) {
        Vt[(hd0 + j) * 64 + key]     = a0[j];
        Vt[(hd0 + 8 + j) * 64 + key] = a1[j];
      }
    }
    wait_async_copies();
    __syncthreads();

    if (kbase <= q0 + 31) {
      // S = Q K^T over hd (2 wmma k-steps), 2x4 frags of 16x16
      v8f sc[2][4];
#pragma unroll
      for (int mf = 0; mf < 2; mf++)
#pragma unroll
        for (int nf = 0; nf < 4; nf++) sc[mf][nf] = v8f_zero();
#pragma unroll
      for (int kc = 0; kc < 2; kc++)
#pragma unroll
        for (int nf = 0; nf < 4; nf++) {
          v16bf kf = load_frag16(&Ks[(nf * 16 + l16) * 64 + kc * 32 + hiq * 8]);
#pragma unroll
          for (int mf = 0; mf < 2; mf++)
            sc[mf][nf] = wmma_bf16(qf[mf][kc], kf, sc[mf][nf]);
        }
      // scale + causal mask + online softmax (row = mf*16 + hiq*8 + r)
#pragma unroll
      for (int mf = 0; mf < 2; mf++)
#pragma unroll
        for (int r = 0; r < 8; r++) {
          const int q = q0 + mf * 16 + hiq * 8 + r;
          float mx = -3.0e38f;
#pragma unroll
          for (int nf = 0; nf < 4; nf++) {
            const int k = kbase + nf * 16 + l16;
            float sv = (k <= q) ? sc[mf][nf][r] * 0.125f : -3.0e38f;
            sc[mf][nf][r] = sv;
            mx = fmaxf(mx, sv);
          }
#pragma unroll
          for (int off = 8; off >= 1; off >>= 1) mx = fmaxf(mx, __shfl_xor(mx, off, 32));
          const float mold = mrow[mf][r];
          const float mnew = fmaxf(mold, mx);
          float alpha = 1.f, rs = 0.f;
          if (mnew > -1.0e37f) {
            alpha = __expf(mold - mnew);
#pragma unroll
            for (int nf = 0; nf < 4; nf++) {
              float p = __expf(sc[mf][nf][r] - mnew);
              sc[mf][nf][r] = p;
              rs += p;
            }
          } else {
#pragma unroll
            for (int nf = 0; nf < 4; nf++) sc[mf][nf][r] = 0.f;
          }
#pragma unroll
          for (int off = 8; off >= 1; off >>= 1) rs += __shfl_xor(rs, off, 32);
          mrow[mf][r] = mnew;
          lrow[mf][r] = lrow[mf][r] * alpha + rs;
#pragma unroll
          for (int nf = 0; nf < 4; nf++) ov[mf][nf][r] = ov[mf][nf][r] * alpha;
        }
      // P: C-layout -> LDS (row-major q x key) -> reload as A-fragments
#pragma unroll
      for (int mf = 0; mf < 2; mf++)
#pragma unroll
        for (int nf = 0; nf < 4; nf++)
#pragma unroll
          for (int r = 0; r < 8; r++)
            Ps[wid][(mf * 16 + hiq * 8 + r) * 64 + nf * 16 + l16] = f2bf(sc[mf][nf][r]);
      // O += P V  (K = 64 keys, 2 wmma k-steps; B-frags from Vt (hd,key))
#pragma unroll
      for (int kc = 0; kc < 2; kc++) {
        v16bf pf[2];
#pragma unroll
        for (int mf = 0; mf < 2; mf++)
          pf[mf] = load_frag16(&Ps[wid][(mf * 16 + l16) * 64 + kc * 32 + hiq * 8]);
#pragma unroll
        for (int nf = 0; nf < 4; nf++) {
          v16bf vf = load_frag16(&Vt[(nf * 16 + l16) * 64 + kc * 32 + hiq * 8]);
#pragma unroll
          for (int mf = 0; mf < 2; mf++)
            ov[mf][nf] = wmma_bf16(pf[mf], vf, ov[mf][nf]);
        }
      }
    }
    __syncthreads();
  }

  // normalize and write bf16 attention rows (b,s, h*64+hd layout == attn_cat)
#pragma unroll
  for (int mf = 0; mf < 2; mf++)
#pragma unroll
    for (int r = 0; r < 8; r++) {
      const int q = q0 + mf * 16 + hiq * 8 + r;
      const float inv = 1.f / fmaxf(lrow[mf][r], 1e-30f);
#pragma unroll
      for (int nf = 0; nf < 4; nf++)
        attnb[(size_t)(b * kS + q) * kD + h * kHD + nf * 16 + l16] =
            f2bf(ov[mf][nf][r] * inv);
    }
}

// ---------------- host launcher ----------------
extern "C" void kernel_launch(void* const* d_in, const int* in_sizes, int n_in,
                              void* d_out, int out_size, void* d_ws, size_t ws_size,
                              hipStream_t stream) {
  (void)in_sizes; (void)n_in; (void)out_size; (void)ws_size;
  const float* x     = (const float*)d_in[0];
  const float* w_qkv = (const float*)d_in[1];
  const float* w_out = (const float*)d_in[2];
  const float* gain  = (const float*)d_in[3];

  float* attn_out = (float*)d_out;                 // B*S*D
  float* vflat    = attn_out + (size_t)kM * kD;    // H*B*S*HD

  // workspace carve-up (~124 MB)
  __bf16* xb    = (__bf16*)d_ws;
  __bf16* wqkvb = xb    + (size_t)kM * kD;     // 8M bf16
  __bf16* woutb = wqkvb + (size_t)kOP * kD;    // 6M
  __bf16* qnb   = woutb + (size_t)kD * kD;     // 4M
  __bf16* knb   = qnb   + (size_t)kM * kD;     // 8M
  __bf16* vbb   = knb   + (size_t)kM * kKVO;   // 2M
  __bf16* atb   = vbb   + (size_t)kM * kKVO;   // 2M
  float*  proj  = (float*)(atb + (size_t)kM * kD);  // 12M f32

  cvt_bf16<<<(kM * kD) / 2048, 256, 0, stream>>>(x, xb, kM * kD);
  cvt_bf16<<<(kOP * kD) / 2048, 256, 0, stream>>>(w_qkv, wqkvb, kOP * kD);
  cvt_bf16<<<(kD * kD) / 2048, 256, 0, stream>>>(w_out, woutb, kD * kD);

  gemm_bf16_nt<<<dim3(kOP / 128, kM / 128), 256, 0, stream>>>(xb, wqkvb, proj,
                                                              kM, kOP, kD);
  rope_norm<<<kM, 256, 0, stream>>>(proj, gain, qnb, knb, vbb, vflat);
  attn_fwd<<<dim3(kS / 256, kH, kB), 256, 0, stream>>>(qnb, knb, vbb, atb);
  gemm_bf16_nt<<<dim3(kD / 128, kM / 128), 256, 0, stream>>>(atb, woutb, attn_out,
                                                             kM, kD, kD);
}
